// Block_64536178590123
// MI455X (gfx1250) — compile-verified, matrix operand order checked by reading, not run on hardware
//
#include <hip/hip_runtime.h>
#include <cstdint>

// ---------------------------------------------------------------------------
// Feature detection for CDNA5 data-movement paths (device pass only)
// ---------------------------------------------------------------------------
#if defined(__HIP_DEVICE_COMPILE__) && defined(__gfx1250__)
#  if __has_builtin(__builtin_amdgcn_tensor_load_to_lds) && \
      __has_builtin(__builtin_amdgcn_s_wait_tensorcnt)
#    define ATH_USE_TDM 1
#    if __clang_major__ >= 23
#      define ATH_TDM_6ARG 1
#    endif
#  elif __has_builtin(__builtin_amdgcn_global_load_async_to_lds_b128) && \
        __has_builtin(__builtin_amdgcn_s_wait_asynccnt)
#    define ATH_USE_ASYNC 1
#  endif
#endif

// ---------------------------------------------------------------------------
// Types for CDNA5 WMMA (gfx1250, wave32)
// ---------------------------------------------------------------------------
typedef __attribute__((ext_vector_type(8)))  __bf16 bf16x8;
typedef __attribute__((ext_vector_type(16))) __bf16 bf16x16;
typedef __attribute__((ext_vector_type(8)))  float  f32x8;

// gcc-style vectors to match clang builtin prototypes exactly
typedef int          tdm_i32x4 __attribute__((vector_size(16)));
typedef int          tdm_i32x8 __attribute__((vector_size(32)));
typedef unsigned int tdm_u32x4 __attribute__((vector_size(16)));

// async-to-LDS builtin takes pointers to vector_size(16) int in AS1/AS3
typedef int ath_v4i __attribute__((vector_size(16)));
typedef __attribute__((address_space(1))) ath_v4i* ath_gptr;
typedef __attribute__((address_space(3))) ath_v4i* ath_lptr;

__device__ __forceinline__ f32x8 wmma_bf16(bf16x16 a, bf16x16 b, f32x8 c) {
  // D = A(16x32 bf16) * B(32x16 bf16) + C(16x16 f32)
  return __builtin_amdgcn_wmma_f32_16x16x32_bf16(
      /*neg_a=*/false, a, /*neg_b=*/false, b,
      /*c_mod=*/(short)0, c, /*reuse_a=*/false, /*reuse_b=*/false);
}

// 16-bit A-matrix 16x32 fragment (also used for B when we have Bt[N,K] N-major):
// lane L: row = L&15, h = L>>4; elems 0..7 = [k0 + h*8 .. +7], 8..15 = [k0+16+h*8 .. +7]
__device__ __forceinline__ bf16x16 load_frag_rm(const __bf16* base, int ld, int k0) {
  const int lane = threadIdx.x & 31;
  const int row  = lane & 15;
  const int h    = lane >> 4;
  const __bf16* p = base + (size_t)row * ld + k0 + h * 8;
  bf16x8 lo = *(const bf16x8*)(p);
  bf16x8 hi = *(const bf16x8*)(p + 16);
  return __builtin_shufflevector(lo, hi, 0,1,2,3,4,5,6,7,8,9,10,11,12,13,14,15);
}

// B-matrix 32x16 fragment from K-major memory (e.g. V[key][d]): lane gives N column,
// elements stride over K rows.
__device__ __forceinline__ bf16x16 load_frag_cm(const __bf16* base, int ld, int n0) {
  const int lane = threadIdx.x & 31;
  const int n = n0 + (lane & 15);
  const int h = lane >> 4;
  bf16x16 r;
#pragma unroll
  for (int j = 0; j < 8; ++j) {
    r[j]     = base[(size_t)(h * 8 + j) * ld + n];
    r[j + 8] = base[(size_t)(16 + h * 8 + j) * ld + n];
  }
  return r;
}

#if defined(ATH_USE_TDM)
// Generic LDS address truncates to the 32-bit LDS offset (ISA: LDS aperture
// addresses map to LDS by discarding upper bits).
__device__ __forceinline__ unsigned lds_offset_of(const void* p) {
  return (unsigned)(size_t)p;
}

// Issue one TDM DMA of a 128-row x 32-col bf16 tile (row stride ldK elements,
// rows contiguous in LDS) from global to LDS. Tracked by TENSORcnt.
__device__ __forceinline__ void tdm_load_tile_128x32_bf16(const __bf16* gsrc,
                                                          int ldK,
                                                          unsigned lds_byte_off) {
  const unsigned long long ga = (unsigned long long)(size_t)gsrc;
  tdm_u32x4 g0;
  g0[0] = 1u;                                                // count=1, user desc
  g0[1] = lds_byte_off;                                      // lds_addr (bytes)
  g0[2] = (unsigned)ga;                                      // global_addr[31:0]
  g0[3] = (unsigned)((ga >> 32) & 0x1FFFFFFu) | (2u << 30);  // addr[56:32] | type=2

  const unsigned td0 = (unsigned)ldK;      // tensor_dim0 (elements)
  const unsigned td1 = 1u << 20;           // tensor_dim1 (generous, OOB only)
  const unsigned st0 = (unsigned)ldK;      // tensor_dim0_stride (elements)
  tdm_i32x8 g1;
  g1[0] = (int)(1u << 16);                                   // mask=0, data_size=2B
  g1[1] = (int)((td0 & 0xFFFFu) << 16);                      // barrier=0 | dim0.lo
  g1[2] = (int)(((td0 >> 16) & 0xFFFFu) | ((td1 & 0xFFFFu) << 16));
  g1[3] = (int)(((td1 >> 16) & 0xFFFFu) | (32u << 16));      // tile_dim0 = 32
  g1[4] = (int)128u;                                         // tile_dim1=128, dim2=0
  g1[5] = (int)st0;                                          // stride0[31:0]
  g1[6] = 0;                                                 // stride0.hi | stride1.lo
  g1[7] = 0;                                                 // stride1.hi
  tdm_i32x4 z4 = {};
#if defined(ATH_TDM_6ARG)
  tdm_i32x8 z8 = {};
  __builtin_amdgcn_tensor_load_to_lds(g0, g1, z4, z4, z8, 0);  // 2D (g2/g3 null)
#else
  __builtin_amdgcn_tensor_load_to_lds(g0, g1, z4, z4, 0);      // 2D (g2/g3 null)
#endif
}
#endif

// ---------------------------------------------------------------------------
// Weight convert + transpose: src f32 [R, Cc] row-major -> dst bf16 [Cc, R]
// batched over blockIdx.z with stride R*Cc.
// ---------------------------------------------------------------------------
__global__ __launch_bounds__(256) void transpose_f32_to_bf16(
    const float* __restrict__ src, __bf16* __restrict__ dst, int R, int Cc) {
  __shared__ float tile[32][33];
  const size_t boff = (size_t)blockIdx.z * R * Cc;
  const float* s = src + boff;
  __bf16* d = dst + boff;
  const int c0 = blockIdx.x * 32, r0 = blockIdx.y * 32;
  const int tx = threadIdx.x & 31, ty = threadIdx.x >> 5;
#pragma unroll
  for (int i = ty; i < 32; i += 8)
    tile[i][tx] = s[(size_t)(r0 + i) * Cc + c0 + tx];
  __syncthreads();
#pragma unroll
  for (int i = ty; i < 32; i += 8)
    d[(size_t)(c0 + i) * R + r0 + tx] = (__bf16)tile[tx][i];
}

// ---------------------------------------------------------------------------
// RMSNorm (C=1024), f32 in -> bf16 out
// ---------------------------------------------------------------------------
__global__ __launch_bounds__(256) void rmsnorm_bf16_kernel(
    const float* __restrict__ x, const float* __restrict__ w,
    __bf16* __restrict__ out) {
  const int row = blockIdx.x;
  const float* xr = x + (size_t)row * 1024;
  const int base = threadIdx.x * 4;
  float v[4];
  float ss = 0.f;
#pragma unroll
  for (int j = 0; j < 4; ++j) { v[j] = xr[base + j]; ss += v[j] * v[j]; }
#pragma unroll
  for (int off = 16; off >= 1; off >>= 1) ss += __shfl_xor(ss, off, 32);
  __shared__ float red[8];
  if ((threadIdx.x & 31) == 0) red[threadIdx.x >> 5] = ss;
  __syncthreads();
  float tot = 0.f;
#pragma unroll
  for (int i = 0; i < 8; ++i) tot += red[i];
  const float rs = rsqrtf(tot * (1.0f / 1024.0f) + 1e-6f);
  __bf16* orow = out + (size_t)row * 1024;
#pragma unroll
  for (int j = 0; j < 4; ++j) orow[base + j] = (__bf16)(v[j] * rs * w[base + j]);
}

// ---------------------------------------------------------------------------
// Tiled bf16 WMMA GEMM:  C[M,N] = A[M,K] * Bt[N,K]^T   (f32 accumulate)
// Block: 256 threads (8 waves), tile 128x128x32. Wave tile 32x64 (2x4 wmma).
// LDS staging via TDM (double-buffered) > async-to-LDS > sync copy.
// EPI: 0 -> f32 store, 1 -> bf16 store, 2 -> exact GELU + bf16 store,
//      3 -> f32 store of acc + resid
// ---------------------------------------------------------------------------
template <int EPI>
__global__ __launch_bounds__(256) void gemm_bf16_kernel(
    const __bf16* __restrict__ A, const __bf16* __restrict__ Bt,
    float* __restrict__ Cf, __bf16* __restrict__ Cb,
    const float* __restrict__ resid, int M, int N, int K) {
  __shared__ __align__(16) __bf16 As[2][128 * 32];
  __shared__ __align__(16) __bf16 Bs[2][128 * 32];
  const int m0 = blockIdx.y * 128, n0 = blockIdx.x * 128;
  const int tid = threadIdx.x, lane = tid & 31, w = tid >> 5;
  const int wm = (w >> 1) * 32, wn = (w & 1) * 64;
  const int nk = K >> 5;

  f32x8 acc[2][4] = {};

#if defined(ATH_USE_TDM)
  // ---- Tensor Data Mover path: one DMA per tile, double buffered ----
  if (tid < 32) {  // wave 0 issues; TDM ignores EXEC, other waves branch around
    tdm_load_tile_128x32_bf16(&A[(size_t)m0 * K], K, lds_offset_of(&As[0][0]));
    tdm_load_tile_128x32_bf16(&Bt[(size_t)n0 * K], K, lds_offset_of(&Bs[0][0]));
  }
  for (int i = 0; i < nk; ++i) {
    const int cur = i & 1;
    if (i + 1 < nk) {
      if (tid < 32) {
        tdm_load_tile_128x32_bf16(&A[(size_t)m0 * K + (i + 1) * 32], K,
                                  lds_offset_of(&As[cur ^ 1][0]));
        tdm_load_tile_128x32_bf16(&Bt[(size_t)n0 * K + (i + 1) * 32], K,
                                  lds_offset_of(&Bs[cur ^ 1][0]));
      }
      __builtin_amdgcn_s_wait_tensorcnt(2);  // tile i complete (in-order per wave)
    } else {
      __builtin_amdgcn_s_wait_tensorcnt(0);
    }
    __syncthreads();
    bf16x16 af[2], bfr[4];
#pragma unroll
    for (int im = 0; im < 2; ++im)
      af[im] = load_frag_rm(&As[cur][(wm + im * 16) * 32], 32, 0);
#pragma unroll
    for (int in = 0; in < 4; ++in)
      bfr[in] = load_frag_rm(&Bs[cur][(wn + in * 16) * 32], 32, 0);
#pragma unroll
    for (int im = 0; im < 2; ++im)
#pragma unroll
      for (int in = 0; in < 4; ++in)
        acc[im][in] = wmma_bf16(af[im], bfr[in], acc[im][in]);
    __syncthreads();
  }
#else
  // ---- async-to-LDS / synchronous staging path (single buffer) ----
  for (int i = 0; i < nk; ++i) {
    const int k0 = i * 32;
    __syncthreads();
#pragma unroll
    for (int c = tid; c < 512; c += 256) {
      const int row = c >> 2, col = (c & 3) * 8;
      const __bf16* ga = &A[(size_t)(m0 + row) * K + k0 + col];
      const __bf16* gb = &Bt[(size_t)(n0 + row) * K + k0 + col];
      __bf16* la = &As[0][row * 32 + col];
      __bf16* lb = &Bs[0][row * 32 + col];
#if defined(ATH_USE_ASYNC)
      __builtin_amdgcn_global_load_async_to_lds_b128(
          (ath_gptr)(size_t)ga, (ath_lptr)(size_t)la, 0, 0);
      __builtin_amdgcn_global_load_async_to_lds_b128(
          (ath_gptr)(size_t)gb, (ath_lptr)(size_t)lb, 0, 0);
#else
      *(uint4*)la = *(const uint4*)ga;
      *(uint4*)lb = *(const uint4*)gb;
#endif
    }
#if defined(ATH_USE_ASYNC)
    __builtin_amdgcn_s_wait_asynccnt(0);
#endif
    __syncthreads();
    if (k0 + 32 < K) {  // global_prefetch_b8 for the next K slice
      __builtin_prefetch(&A[(size_t)(m0 + (tid >> 1)) * K + k0 + 32 + (tid & 1) * 16], 0, 1);
      __builtin_prefetch(&Bt[(size_t)(n0 + (tid >> 1)) * K + k0 + 32 + (tid & 1) * 16], 0, 1);
    }
    bf16x16 af[2], bfr[4];
#pragma unroll
    for (int im = 0; im < 2; ++im)
      af[im] = load_frag_rm(&As[0][(wm + im * 16) * 32], 32, 0);
#pragma unroll
    for (int in = 0; in < 4; ++in)
      bfr[in] = load_frag_rm(&Bs[0][(wn + in * 16) * 32], 32, 0);
#pragma unroll
    for (int im = 0; im < 2; ++im)
#pragma unroll
      for (int in = 0; in < 4; ++in)
        acc[im][in] = wmma_bf16(af[im], bfr[in], acc[im][in]);
  }
#endif

  // Epilogue. C frag: lane n = L&15, h = L>>4; VGPR r -> row r + 8h.
  const int n_ = lane & 15, hh = lane >> 4;
#pragma unroll
  for (int im = 0; im < 2; ++im)
#pragma unroll
    for (int in = 0; in < 4; ++in)
#pragma unroll
      for (int r = 0; r < 8; ++r) {
        const int gm = m0 + wm + im * 16 + r + 8 * hh;
        const int gn = n0 + wn + in * 16 + n_;
        const size_t idx = (size_t)gm * N + gn;
        float v = acc[im][in][r];
        if (EPI == 1) {
          Cb[idx] = (__bf16)v;
        } else if (EPI == 2) {
          v = 0.5f * v * (1.0f + erff(v * 0.70710678118654752f));
          Cb[idx] = (__bf16)v;
        } else if (EPI == 3) {
          Cf[idx] = v + resid[idx];
        } else {
          Cf[idx] = v;
        }
      }
}

// ---------------------------------------------------------------------------
// Flash attention, causal. qkv bf16 [B*T, 3072] (q|k|v, head h at col h*64).
// Block = 128 threads (4 waves); each wave owns 32 queries, online softmax.
// ---------------------------------------------------------------------------
__global__ __launch_bounds__(128) void flash_attn_kernel(
    const __bf16* __restrict__ qkv, __bf16* __restrict__ y) {
  __shared__ __align__(16) __bf16 Plds[4][32][32];
  const int wave = threadIdx.x >> 5, lane = threadIdx.x & 31;
  const int nl = lane & 15, hl = lane >> 4;
  const int bh = blockIdx.y;
  const int b = bh >> 4, h = bh & 15;
  const int q0 = blockIdx.x * 128 + wave * 32;
  const size_t bbase = (size_t)b * 2048 * 3072;
  const __bf16* Qp = qkv + bbase + h * 64;
  const __bf16* Kp = qkv + bbase + 1024 + h * 64;
  const __bf16* Vp = qkv + bbase + 2048 + h * 64;

  bf16x16 qf[2][2];
#pragma unroll
  for (int im = 0; im < 2; ++im)
#pragma unroll
    for (int kk = 0; kk < 2; ++kk)
      qf[im][kk] = load_frag_rm(Qp + (size_t)(q0 + im * 16) * 3072, 3072, kk * 32);

  f32x8 o[2][4] = {};
  float Mst[2][8], Lst[2][8];
#pragma unroll
  for (int im = 0; im < 2; ++im)
#pragma unroll
    for (int r = 0; r < 8; ++r) { Mst[im][r] = -1e30f; Lst[im][r] = 0.f; }

  const int nkt = (q0 >> 5) + 1;  // causal: key tiles 0..q0/32
  for (int kt = 0; kt < nkt; ++kt) {
    const int t0 = kt * 32;
    bf16x16 kf[2][2];
#pragma unroll
    for (int in = 0; in < 2; ++in)
#pragma unroll
      for (int kk = 0; kk < 2; ++kk)
        kf[in][kk] = load_frag_rm(Kp + (size_t)(t0 + in * 16) * 3072, 3072, kk * 32);

    f32x8 s[2][2] = {};
#pragma unroll
    for (int im = 0; im < 2; ++im)
#pragma unroll
      for (int in = 0; in < 2; ++in) {
        s[im][in] = wmma_bf16(qf[im][0], kf[in][0], s[im][in]);
        s[im][in] = wmma_bf16(qf[im][1], kf[in][1], s[im][in]);
      }

    // scale + causal mask
#pragma unroll
    for (int im = 0; im < 2; ++im)
#pragma unroll
      for (int in = 0; in < 2; ++in)
#pragma unroll
        for (int r = 0; r < 8; ++r) {
          const int qi = q0 + im * 16 + r + 8 * hl;
          const int ki = t0 + in * 16 + nl;
          float v = s[im][in][r] * 0.125f;  // 1/sqrt(64)
          if (ki > qi) v = -3.0e38f;
          s[im][in][r] = v;
        }

    // online softmax (row spans 16 lanes of a half-wave)
#pragma unroll
    for (int im = 0; im < 2; ++im)
#pragma unroll
      for (int r = 0; r < 8; ++r) {
        float t = fmaxf(s[im][0][r], s[im][1][r]);
#pragma unroll
        for (int off = 1; off < 16; off <<= 1) t = fmaxf(t, __shfl_xor(t, off, 32));
        const float mo = Mst[im][r];
        const float mn = fmaxf(mo, t);
        const float alpha = __expf(mo - mn);
        const float p0 = __expf(s[im][0][r] - mn);
        const float p1 = __expf(s[im][1][r] - mn);
        float ps = p0 + p1;
#pragma unroll
        for (int off = 1; off < 16; off <<= 1) ps += __shfl_xor(ps, off, 32);
        Lst[im][r] = Lst[im][r] * alpha + ps;
        Mst[im][r] = mn;
#pragma unroll
        for (int dn = 0; dn < 4; ++dn) o[im][dn][r] = o[im][dn][r] * alpha;
        Plds[wave][im * 16 + r + 8 * hl][nl]      = (__bf16)p0;
        Plds[wave][im * 16 + r + 8 * hl][16 + nl] = (__bf16)p1;
      }

    // P @ V  (same-wave LDS store->load; DS ops are in-order within a wave)
    bf16x16 pa[2];
#pragma unroll
    for (int im = 0; im < 2; ++im)
      pa[im] = load_frag_rm(&Plds[wave][im * 16][0], 32, 0);
#pragma unroll
    for (int dn = 0; dn < 4; ++dn) {
      bf16x16 vb = load_frag_cm(Vp + (size_t)t0 * 3072, 3072, dn * 16);
#pragma unroll
      for (int im = 0; im < 2; ++im) o[im][dn] = wmma_bf16(pa[im], vb, o[im][dn]);
    }
  }

  // write y (bf16 [B*T, 1024])
#pragma unroll
  for (int im = 0; im < 2; ++im)
#pragma unroll
    for (int dn = 0; dn < 4; ++dn)
#pragma unroll
      for (int r = 0; r < 8; ++r) {
        const int q = q0 + im * 16 + r + 8 * hl;
        const float denom = Lst[im][r];
        y[(size_t)(b * 2048 + q) * 1024 + h * 64 + dn * 16 + nl] =
            (__bf16)(o[im][dn][r] / denom);
      }
}

// ---------------------------------------------------------------------------
// Router: logits[n, e] = xn2[n, :] . Wr[:, e]  (E=8), + softmax probs
// ---------------------------------------------------------------------------
__global__ __launch_bounds__(256) void router_kernel(
    const __bf16* __restrict__ xn2, const float* __restrict__ Wr,
    float* __restrict__ logits, float* __restrict__ probs) {
  const int n = blockIdx.x;
  const int tid = threadIdx.x;
  const int e = tid & 7, chunk = tid >> 3;
  const __bf16* xr = xn2 + (size_t)n * 1024;
  float p = 0.f;
#pragma unroll 8
  for (int j = 0; j < 32; ++j) {
    const int c = chunk * 32 + j;
    p += (float)xr[c] * Wr[c * 8 + e];
  }
  __shared__ float red[256];
  red[tid] = p;
  __syncthreads();
  for (int s = 128; s >= 8; s >>= 1) {
    if (tid < s) red[tid] += red[tid + s];
    __syncthreads();
  }
  if (tid < 8) {
    logits[(size_t)n * 8 + tid] = red[tid];
    float mx = red[0];
#pragma unroll
    for (int i = 1; i < 8; ++i) mx = fmaxf(mx, red[i]);
    float sum = 0.f;
#pragma unroll
    for (int i = 0; i < 8; ++i) sum += __expf(red[i] - mx);
    probs[(size_t)n * 8 + tid] = __expf(red[tid] - mx) / sum;
  }
}

// ---------------------------------------------------------------------------
// Expert-choice top-k (k=2048 of N=8192) via in-LDS bitonic sort of packed keys.
// key = (prob_bits & ~0x1FFF) | (8191 - idx); sort ascending on ~key
// ---------------------------------------------------------------------------
__global__ __launch_bounds__(1024) void topk_kernel(
    const float* __restrict__ probs, int* __restrict__ selected,
    float* __restrict__ weights) {
  const int e = blockIdx.x;
  const int tid = threadIdx.x;
  __shared__ unsigned keys[8192];
  for (int i = tid; i < 8192; i += 1024) {
    const unsigned b = __float_as_uint(probs[(size_t)i * 8 + e]);
    keys[i] = ~((b & 0xFFFFE000u) | (unsigned)(8191 - i));
  }
  __syncthreads();
  for (int k = 2; k <= 8192; k <<= 1) {
    for (int j = k >> 1; j > 0; j >>= 1) {
      for (int i = tid; i < 8192; i += 1024) {
        const int ixj = i ^ j;
        if (ixj > i) {
          const bool up = ((i & k) == 0);
          const unsigned a = keys[i], c = keys[ixj];
          if ((a > c) == up) { keys[i] = c; keys[ixj] = a; }
        }
      }
      __syncthreads();
    }
  }
  for (int i = tid; i < 2048; i += 1024) {
    const unsigned kk = ~keys[i];
    const int n = 8191 - (int)(kk & 0x1FFFu);
    selected[e * 2048 + i] = n;
    weights[e * 2048 + i] = probs[(size_t)n * 8 + e];
  }
}

// ---------------------------------------------------------------------------
// Gather selected token rows (bf16, C=1024 -> 128 uint4)
// ---------------------------------------------------------------------------
__global__ __launch_bounds__(128) void gather_kernel(
    const __bf16* __restrict__ xn2, const int* __restrict__ selected,
    __bf16* __restrict__ tokens) {
  const int row = blockIdx.x;
  const int n = selected[row];
  ((uint4*)(tokens + (size_t)row * 1024))[threadIdx.x] =
      ((const uint4*)(xn2 + (size_t)n * 1024))[threadIdx.x];
}

// ---------------------------------------------------------------------------
// Weighted scatter-add of one expert's outputs
// ---------------------------------------------------------------------------
__global__ __launch_bounds__(256) void scatter_kernel(
    const float* __restrict__ moe_out, const int* __restrict__ selected,
    const float* __restrict__ weights, float* __restrict__ accum) {
  const int i = blockIdx.x;
  const int n = selected[i];
  const float w = weights[i];
  for (int c = threadIdx.x; c < 1024; c += 256)
    atomicAdd(&accum[(size_t)n * 1024 + c], w * moe_out[(size_t)i * 1024 + c]);
}

__global__ __launch_bounds__(256) void final_add_kernel(
    const float* __restrict__ x1, const float* __restrict__ accum,
    float* __restrict__ out) {
  const size_t i = (size_t)blockIdx.x * 256 + threadIdx.x;
  out[i] = x1[i] + accum[i];
}

__global__ __launch_bounds__(256) void emit_selected_kernel(
    const int* __restrict__ sel, int* __restrict__ out) {
  const int i = blockIdx.x * 256 + threadIdx.x;
  out[i] = sel[i];
}

// ---------------------------------------------------------------------------
// Host orchestration
// ---------------------------------------------------------------------------
extern "C" void kernel_launch(void* const* d_in, const int* in_sizes, int n_in,
                              void* d_out, int out_size, void* d_ws, size_t ws_size,
                              hipStream_t stream) {
  (void)in_sizes; (void)n_in; (void)out_size; (void)ws_size;
  const float* x      = (const float*)d_in[0];
  const float* ln1_w  = (const float*)d_in[1];
  const float* ln2_w  = (const float*)d_in[2];
  const float* W_attn = (const float*)d_in[3];
  const float* W_o    = (const float*)d_in[4];
  const float* W_r    = (const float*)d_in[5];
  const float* W_fc   = (const float*)d_in[6];
  const float* W_pr   = (const float*)d_in[7];

  constexpr int Bv = 4, Tv = 2048, Cv = 1024, NHv = 16, Ev = 8, HIDv = 4096;
  constexpr int Nrows = Bv * Tv;  // 8192
  constexpr int TOPK = 2048;

  char* wsp = (char*)d_ws;
  size_t off = 0;
  auto alloc = [&](size_t bytes) -> void* {
    off = (off + 255) & ~(size_t)255;
    void* p = wsp + off;
    off += bytes;
    return p;
  };
  __bf16* xn1    = (__bf16*)alloc((size_t)Nrows * Cv * 2);
  __bf16* WattnT = (__bf16*)alloc((size_t)3 * Cv * Cv * 2);
  __bf16* WoT    = (__bf16*)alloc((size_t)Cv * Cv * 2);
  __bf16* qkv    = (__bf16*)alloc((size_t)Nrows * 3 * Cv * 2);
  __bf16* yat    = (__bf16*)alloc((size_t)Nrows * Cv * 2);
  float*  x1     = (float*)alloc((size_t)Nrows * Cv * 4);
  __bf16* xn2    = (__bf16*)alloc((size_t)Nrows * Cv * 2);
  float*  probs  = (float*)alloc((size_t)Nrows * Ev * 4);
  int*    sel    = (int*)alloc((size_t)Ev * TOPK * 4);
  float*  selw   = (float*)alloc((size_t)Ev * TOPK * 4);
  __bf16* WfcT   = (__bf16*)alloc((size_t)Ev * Cv * HIDv * 2);
  __bf16* WprT   = (__bf16*)alloc((size_t)Ev * HIDv * Cv * 2);
  __bf16* tokens = (__bf16*)alloc((size_t)Ev * TOPK * Cv * 2);
  __bf16* hbuf   = (__bf16*)alloc((size_t)TOPK * HIDv * 2);
  float*  moebuf = (float*)alloc((size_t)TOPK * Cv * 4);
  float*  accum  = (float*)alloc((size_t)Nrows * Cv * 4);

  float* out_x      = (float*)d_out;
  float* out_logits = out_x + (size_t)Nrows * Cv;
  int*   out_sel    = (int*)(out_logits + (size_t)Nrows * Ev);

  // Weight conversion (f32 [K,N] -> bf16 [N,K])
  transpose_f32_to_bf16<<<dim3(3 * Cv / 32, Cv / 32, 1), 256, 0, stream>>>(W_attn, WattnT, Cv, 3 * Cv);
  transpose_f32_to_bf16<<<dim3(Cv / 32, Cv / 32, 1), 256, 0, stream>>>(W_o, WoT, Cv, Cv);
  transpose_f32_to_bf16<<<dim3(HIDv / 32, Cv / 32, Ev), 256, 0, stream>>>(W_fc, WfcT, Cv, HIDv);
  transpose_f32_to_bf16<<<dim3(Cv / 32, HIDv / 32, Ev), 256, 0, stream>>>(W_pr, WprT, HIDv, Cv);

  // Attention path
  rmsnorm_bf16_kernel<<<Nrows, 256, 0, stream>>>(x, ln1_w, xn1);
  gemm_bf16_kernel<1><<<dim3(3 * Cv / 128, Nrows / 128), 256, 0, stream>>>(
      xn1, WattnT, nullptr, qkv, nullptr, Nrows, 3 * Cv, Cv);
  flash_attn_kernel<<<dim3(Tv / 128, Bv * NHv), 128, 0, stream>>>(qkv, yat);
  gemm_bf16_kernel<3><<<dim3(Cv / 128, Nrows / 128), 256, 0, stream>>>(
      yat, WoT, x1, nullptr, x, Nrows, Cv, Cv);

  // MoE path
  rmsnorm_bf16_kernel<<<Nrows, 256, 0, stream>>>(x1, ln2_w, xn2);
  router_kernel<<<Nrows, 256, 0, stream>>>(xn2, W_r, out_logits, probs);
  topk_kernel<<<Ev, 1024, 0, stream>>>(probs, sel, selw);
  gather_kernel<<<Ev * TOPK, 128, 0, stream>>>(xn2, sel, tokens);
  (void)hipMemsetAsync(accum, 0, (size_t)Nrows * Cv * 4, stream);
  for (int e = 0; e < Ev; ++e) {
    gemm_bf16_kernel<2><<<dim3(HIDv / 128, TOPK / 128), 256, 0, stream>>>(
        tokens + (size_t)e * TOPK * Cv, WfcT + (size_t)e * HIDv * Cv,
        nullptr, hbuf, nullptr, TOPK, HIDv, Cv);
    gemm_bf16_kernel<0><<<dim3(Cv / 128, TOPK / 128), 256, 0, stream>>>(
        hbuf, WprT + (size_t)e * Cv * HIDv, moebuf, nullptr, nullptr, TOPK, Cv, HIDv);
    scatter_kernel<<<TOPK, 256, 0, stream>>>(moebuf, sel + e * TOPK, selw + e * TOPK, accum);
  }
  final_add_kernel<<<(Nrows * Cv) / 256, 256, 0, stream>>>(x1, accum, out_x);
  emit_selected_kernel<<<(Ev * TOPK) / 256, 256, 0, stream>>>(sel, out_sel);
}
